// MultiScaleConditionFusion_38096359915876
// MI455X (gfx1250) — matrix operand order repaired and checked by reading, not compiled, not run on hardware
//
#include <hip/hip_runtime.h>

typedef __attribute__((ext_vector_type(16))) __bf16 v16bf;
typedef __attribute__((ext_vector_type(8)))  float  v8f;
typedef __attribute__((ext_vector_type(4)))  float  f32x4;
typedef __attribute__((ext_vector_type(4)))  unsigned int u32x4;

namespace {
constexpr int kB  = 32;
constexpr int kS  = 2048;
constexpr int kD  = 1024;
constexpr int BM  = 128;
constexpr int BN  = 128;
constexpr int BK  = 32;
constexpr int LSTR = 40;          // bf16 units per LDS row (80B pitch -> conflict-free)
constexpr int NMT  = kS / BM;     // 16 m-tiles per batch
}

// f32 -> bf16 round-to-nearest (half-up), two packed per dword:
// 2x v_add + 1x v_perm_b32 = 1.5 VALU ops per element pair half,
// vs 3 ops/element for the bfe/add3/mov RNE sequence.
__device__ __forceinline__ unsigned int packbf2(float lo, float hi) {
#if __has_builtin(__builtin_amdgcn_perm)
  unsigned int a = __builtin_bit_cast(unsigned int, lo) + 0x8000u;
  unsigned int b = __builtin_bit_cast(unsigned int, hi) + 0x8000u;
  // v_perm_b32: bytes 0-3 from 2nd src (a), 4-7 from 1st src (b);
  // dest = [b.hi16 | a.hi16]
  return __builtin_amdgcn_perm(b, a, 0x07060302u);
#else
  unsigned int a = __builtin_bit_cast(unsigned int, lo);
  a += 0x7FFFu + ((a >> 16) & 1u);
  unsigned int b = __builtin_bit_cast(unsigned int, hi);
  b += 0x7FFFu + ((b >> 16) & 1u);
  return (a >> 16) | (b & 0xFFFF0000u);
#endif
}

__device__ __forceinline__ u32x4 pack8(f32x4 a, f32x4 b) {
  u32x4 r;
  r.x = packbf2(a.x, a.y);
  r.y = packbf2(a.z, a.w);
  r.z = packbf2(b.x, b.y);
  r.w = packbf2(b.z, b.w);
  return r;
}

// Fused: Y = silu(X @ W[sid]^T + b[sid]) + scale_embed[sid];
// high = Y + be0, mid = Y + be1; deterministic masked column partials -> ws.
__global__ void __launch_bounds__(256)
msc_fused_gemm(const float* __restrict__ X,            // (B,S,D)
               const int*   __restrict__ scale_id,     // (B)
               const int*   __restrict__ midi_mask,    // (B,S)
               const float* __restrict__ W,            // (3,D,D)
               const float* __restrict__ bias,         // (3,D)
               const float* __restrict__ scale_embed,  // (3,D)
               const float* __restrict__ branch_embed, // (3,D)
               float* __restrict__ out_high,           // (B,S,D)
               float* __restrict__ out_mid,            // (B,S,D)
               float* __restrict__ ws_partial)         // (B,NMT,D)
{
  __shared__ __align__(16) unsigned short As[2][BM * LSTR];
  __shared__ __align__(16) unsigned short Bs[2][BN * LSTR];
  __shared__ float maskf[BM];
  __shared__ float partLds[8][4][32];

  const int n0 = blockIdx.x * BN;
  const int m0 = blockIdx.y * BM;
  const int bz = blockIdx.z;

  int sid = scale_id[bz];
  sid = sid < 0 ? 0 : (sid > 2 ? 2 : sid);

  const int t     = threadIdx.x;
  const int lane  = t & 31;
  const int wave  = t >> 5;
  const int waveM = wave >> 1;          // 0..3 -> 32-row strip
  const int waveN = wave & 1;           // 0..1 -> 64-col strip
  const int fragRow = lane & 15;
  const int kHalfA  = (lane >> 4) << 3; // A frag: K base 0 / 8  (ISA 16-bit A layout)
  const int kHalfB  = (lane >> 4) << 3; // B frag: K base 0 / 8  (same layout as A)

  const float* Xb = X + (size_t)bz * kS * kD + (size_t)m0 * kD;
  const float* Wb = W + (size_t)sid * kD * kD + (size_t)n0 * kD;

  const int ldRow = t >> 1;             // 128 rows, 2 threads per row
  const int ldK   = (t & 1) << 4;       // 0 / 16 within 32-wide K slab

  v8f acc[2][4];
#pragma unroll
  for (int i = 0; i < 2; ++i)
#pragma unroll
    for (int j = 0; j < 4; ++j)
      acc[i][j] = (v8f){0.f, 0.f, 0.f, 0.f, 0.f, 0.f, 0.f, 0.f};

  f32x4 ra[4], rb[4];                   // global staging (next K slab)

  auto loadG = [&](int kt) {
    const f32x4* sa = (const f32x4*)(Xb + (size_t)ldRow * kD + kt + ldK);
    ra[0] = sa[0]; ra[1] = sa[1]; ra[2] = sa[2]; ra[3] = sa[3];
    const f32x4* sb = (const f32x4*)(Wb + (size_t)ldRow * kD + kt + ldK);
    rb[0] = sb[0]; rb[1] = sb[1]; rb[2] = sb[2]; rb[3] = sb[3];
  };
  auto storeL = [&](int buf) {
    *(u32x4*)&As[buf][ldRow * LSTR + ldK]     = pack8(ra[0], ra[1]);
    *(u32x4*)&As[buf][ldRow * LSTR + ldK + 8] = pack8(ra[2], ra[3]);
    *(u32x4*)&Bs[buf][ldRow * LSTR + ldK]     = pack8(rb[0], rb[1]);
    *(u32x4*)&Bs[buf][ldRow * LSTR + ldK + 8] = pack8(rb[2], rb[3]);
  };
  auto compute = [&](int buf) {
    union Frag { u32x4 u[2]; v16bf v; };
    Frag fa[2], fb[4];
#pragma unroll
    for (int i = 0; i < 2; ++i) {
      const int r = waveM * 32 + i * 16 + fragRow;               // A row
      fa[i].u[0] = *(const u32x4*)&As[buf][r * LSTR + kHalfA];        // K 0..7  / 8..15
      fa[i].u[1] = *(const u32x4*)&As[buf][r * LSTR + kHalfA + 16];   // K 16..23/ 24..31
    }
#pragma unroll
    for (int j = 0; j < 4; ++j) {
      const int r = waveN * 64 + j * 16 + fragRow;               // W row == B column
      fb[j].u[0] = *(const u32x4*)&Bs[buf][r * LSTR + kHalfB];        // K 0..7  / 8..15
      fb[j].u[1] = *(const u32x4*)&Bs[buf][r * LSTR + kHalfB + 16];   // K 16..23/ 24..31
    }
#pragma unroll
    for (int i = 0; i < 2; ++i)
#pragma unroll
      for (int j = 0; j < 4; ++j)
        acc[i][j] = __builtin_amdgcn_wmma_f32_16x16x32_bf16(
            false, fa[i].v, false, fb[j].v, (short)0, acc[i][j], false, false);
  };

  // ---- software-pipelined main loop: one barrier per K-step, double-buffered LDS
  loadG(0);
  storeL(0);
  __syncthreads();
  int cur = 0;
  for (int kt = BK; kt < kD; kt += BK) {
    loadG(kt);              // prefetch next slab (hidden under WMMA phase)
    compute(cur);
    storeL(cur ^ 1);        // write other buffer: no hazard with readers of `cur`
    __syncthreads();
    cur ^= 1;
  }
  compute(cur);

  // ---- epilogue: bias + SiLU + scale_embed, branch adds, masked column partials
  if (t < BM)
    maskf[t] = (midi_mask[(size_t)bz * kS + m0 + t] > 0) ? 1.0f : 0.0f;
  __syncthreads();

  float* outH = out_high + (size_t)bz * kS * kD;
  float* outM = out_mid  + (size_t)bz * kS * kD;
  const int hi8 = (lane >> 4) << 3;     // C layout: lanes 16-31 hold M = v+8

  float part[4] = {0.f, 0.f, 0.f, 0.f};

#pragma unroll
  for (int j = 0; j < 4; ++j) {
    const int col = n0 + waveN * 64 + j * 16 + fragRow;
    const float bb  = bias[sid * kD + col];
    const float se  = scale_embed[sid * kD + col];
    const float be0 = branch_embed[0 * kD + col];
    const float be1 = branch_embed[1 * kD + col];
#pragma unroll
    for (int i = 0; i < 2; ++i) {
      const int rbase = waveM * 32 + i * 16 + hi8;
#pragma unroll
      for (int v = 0; v < 8; ++v) {
        const int rowLoc = rbase + v;
        const float val = acc[i][j][v] + bb;
        const float y = val * (1.0f / (1.0f + __expf(-val))) + se;  // silu + embed
        const size_t idx = (size_t)(m0 + rowLoc) * kD + col;
        // streamed outputs: never re-read -> non-temporal, keep X/W hot in L2
        __builtin_nontemporal_store(y + be0, &outH[idx]);
        __builtin_nontemporal_store(y + be1, &outM[idx]);
        part[j] += maskf[rowLoc] * y;
      }
    }
  }

  // deterministic cross-wave column reduction (fixed order, no atomics)
#pragma unroll
  for (int j = 0; j < 4; ++j) partLds[wave][j][lane] = part[j];
  __syncthreads();

  if (t < BN) {
    const int wN = t >> 6, jj = (t >> 4) & 3, l15 = t & 15;
    float s = 0.f;
#pragma unroll
    for (int wm = 0; wm < 4; ++wm) {
      const int w = wm * 2 + wN;
      s += partLds[w][jj][l15] + partLds[w][jj][l15 + 16];
    }
    ws_partial[((size_t)bz * NMT + blockIdx.y) * kD + n0 + t] = s;
  }
}

// low_cond = (masked sum over S) / max(count,1e-6) + branch_embed[2]
__global__ void __launch_bounds__(256)
msc_low_cond(const int*   __restrict__ midi_mask,    // (B,S)
             const float* __restrict__ branch_embed, // (3,D)
             const float* __restrict__ ws_partial,   // (B,NMT,D)
             float* __restrict__ out_low)            // (B,D)
{
  __shared__ float red[256];
  const int bz = blockIdx.x;
  const int t  = threadIdx.x;

  float cnt = 0.f;
  for (int s = t; s < kS; s += 256)
    cnt += (midi_mask[(size_t)bz * kS + s] > 0) ? 1.0f : 0.0f;
  red[t] = cnt;
  __syncthreads();
  for (int off = 128; off > 0; off >>= 1) {
    if (t < off) red[t] += red[t + off];
    __syncthreads();
  }
  const float denom = fmaxf(red[0], 1e-6f);

  for (int e = t; e < kD; e += 256) {
    float s = 0.f;
#pragma unroll
    for (int mt = 0; mt < NMT; ++mt)
      s += ws_partial[((size_t)bz * NMT + mt) * kD + e];
    out_low[(size_t)bz * kD + e] = s / denom + branch_embed[2 * kD + e];
  }
}

extern "C" void kernel_launch(void* const* d_in, const int* in_sizes, int n_in,
                              void* d_out, int out_size, void* d_ws, size_t ws_size,
                              hipStream_t stream) {
  const float* X    = (const float*)d_in[0];   // midi_features (B,S,D) f32
  const int*   sid  = (const int*)  d_in[1];   // scale_id (B) i32
  const int*   mask = (const int*)  d_in[2];   // midi_mask (B,S) i32
  const float* W    = (const float*)d_in[3];   // (3,D,D) f32
  const float* bia  = (const float*)d_in[4];   // (3,D) f32
  const float* se   = (const float*)d_in[5];   // (3,D) f32
  const float* be   = (const float*)d_in[6];   // (3,D) f32

  float* out  = (float*)d_out;
  float* outH = out;                                  // (B,S,D)
  float* outM = out + (size_t)kB * kS * kD;           // (B,S,D)
  float* outL = out + (size_t)2 * kB * kS * kD;       // (B,1,D)
  float* ws   = (float*)d_ws;                         // B*NMT*D f32 = 2 MB

  dim3 grid(kD / BN, kS / BM, kB);   // (8, 16, 32)
  msc_fused_gemm<<<grid, 256, 0, stream>>>(X, sid, mask, W, bia, se, be,
                                           outH, outM, ws);
  msc_low_cond<<<kB, 256, 0, stream>>>(mask, be, ws, outL);
}
